// VisionTransformer_53979148976635
// MI455X (gfx1250) — matrix operand order repaired and checked by reading, not compiled, not run on hardware
//
#include <hip/hip_runtime.h>
#include <cstdint>
#include <cstddef>

// ---------------- types ----------------
typedef __attribute__((ext_vector_type(16))) __bf16 v16bf;
typedef __attribute__((ext_vector_type(8)))  float  v8f;

#define N_REAL 2049
#define S_PAD 4096
#define EMBED 768
#define HEADS 12
#define DH 64
#define MLP_DIM 3072

__device__ __constant__ int d_segw[5]  = {128, 256, 512, 1024, 2048};
__device__ __constant__ int d_ratio[5] = {1, 2, 4, 8, 16};
__device__ __constant__ int d_cum[6]   = {0, 32, 48, 56, 60, 62};  // cumulative segments

// ---------------- CDNA5 Tensor Data Mover (guarded) ----------------
#if defined(__has_builtin)
#  if __has_builtin(__builtin_amdgcn_tensor_load_to_lds) && __has_builtin(__builtin_amdgcn_s_wait_tensorcnt)
#    define USE_TDM 1
#  endif
#endif

#ifdef USE_TDM
typedef unsigned int u32x4 __attribute__((ext_vector_type(4)));
typedef int          i32x8 __attribute__((ext_vector_type(8)));
typedef int          i32x4 __attribute__((ext_vector_type(4)));

// 2D bf16 tile load, global -> LDS, with per-row LDS padding.
// tile_w halves per row, tile_h rows, row stride stride_elems (halves).
// pad_interval_code: rows of 2<<code DWORDs; pad_amount_code: (code+1) DWORDs appended.
__device__ inline void tdm_load_2d(const __bf16* gsrc, void* lds_dst,
                                   unsigned tile_w, unsigned tile_h,
                                   unsigned long long stride_elems,
                                   unsigned pad_interval_code, unsigned pad_amount_code) {
    unsigned long long ga = (unsigned long long)(uintptr_t)gsrc;  // byte address
    unsigned lds = (unsigned)(uintptr_t)lds_dst;                  // low 32 bits = LDS offset
    u32x4 g0;
    g0[0] = 1u;                                                    // count=1 (valid), user mode
    g0[1] = lds;                                                   // lds_addr
    g0[2] = (unsigned)(ga & 0xFFFFFFFFu);                          // global_addr[31:0]
    g0[3] = (unsigned)((ga >> 32) & 0x01FFFFFFu) | (2u << 30);     // global_addr[56:32] | type=2
    i32x8 g1;
    g1[0] = (int)((1u << 16)                        // data_size = 1 (2 bytes)
                  | (1u << 20)                      // pad_enable
                  | (pad_interval_code << 22)
                  | (pad_amount_code << 25));
    g1[1] = (int)((tile_w & 0xFFFFu) << 16);                       // tensor_dim0[15:0]
    g1[2] = (int)((tile_w >> 16) | ((tile_h & 0xFFFFu) << 16));    // dim0[31:16] | dim1[15:0]
    g1[3] = (int)((tile_h >> 16) | ((tile_w & 0xFFFFu) << 16));    // dim1[31:16] | tile_dim0
    g1[4] = (int)(tile_h & 0xFFFFu);                               // tile_dim1; tile_dim2=0
    g1[5] = (int)(stride_elems & 0xFFFFFFFFull);                   // tensor_dim0_stride lo
    g1[6] = (int)((stride_elems >> 32) & 0xFFFFull);               // stride hi; dim1_stride=0
    g1[7] = 0;
    i32x4 z;
    z[0] = z[1] = z[2] = z[3] = 0;
    i32x8 z8;
#pragma unroll
    for (int i = 0; i < 8; ++i) z8[i] = 0;
    // 6-arg form (upstream clang-23 / therock-10.0 headers)
    __builtin_amdgcn_tensor_load_to_lds(g0, g1, z, z, z8, 0);
}
__device__ inline void tdm_wait() { __builtin_amdgcn_s_wait_tensorcnt(0); }
#endif

// ---------------- WMMA fragment helpers (CDNA5 16x16x32 bf16 layouts) ----------------
// A matrix 16x32 (16-bit): lane<16 -> M=lane, K = {ko..ko+7, ko+16..ko+23}
//                          lane>=16 -> M=lane-16, K = {ko+8..ko+15, ko+24..ko+31}
__device__ inline v16bf load_a_frag(const __bf16* base, int lda, int row_base, int ko, int lane) {
    int m  = row_base + (lane & 15);
    int kb = ko + ((lane & 16) ? 8 : 0);
    const __bf16* p = base + (size_t)m * lda + kb;
    v16bf a;
#pragma unroll
    for (int i = 0; i < 8; ++i) a[i] = p[i];
#pragma unroll
    for (int i = 0; i < 8; ++i) a[8 + i] = p[16 + i];
    return a;
}
// B matrix 32x16 (16-bit), stored transposed BT[n][k]: lane<16 -> N=lane, K = {ko..ko+7, ko+16..ko+23};
//                          lane>=16 -> N=lane-16, K = {ko+8..ko+15, ko+24..ko+31}.
__device__ inline v16bf load_b_frag(const __bf16* baseT, int ldb, int n_base, int ko, int lane) {
    int n  = n_base + (lane & 15);
    int kb = ko + ((lane & 16) ? 8 : 0);
    const __bf16* p = baseT + (size_t)n * ldb + kb;
    v16bf b;
#pragma unroll
    for (int i = 0; i < 8; ++i) b[i] = p[i];
#pragma unroll
    for (int i = 0; i < 8; ++i) b[8 + i] = p[16 + i];
    return b;
}
__device__ inline v8f wmma_bf16(v16bf a, v16bf b, v8f c) {
    return __builtin_amdgcn_wmma_f32_16x16x32_bf16(false, a, false, b, (short)0, c, false, false);
}
__device__ inline v8f v8f_zero() {
    v8f z;
#pragma unroll
    for (int i = 0; i < 8; ++i) z[i] = 0.f;
    return z;
}

// ---------------- fp32 -> bf16 conversion ----------------
__global__ void cvt_bf16_kernel(const float* __restrict__ src, __bf16* __restrict__ dst, long n) {
    long i = (long)blockIdx.x * 256 + threadIdx.x;
    if (i < n) dst[i] = (__bf16)src[i];
}

// ---------------- h init: cls row + zero pad rows ----------------
__global__ void init_h_kernel(const float* __restrict__ cls, const float* __restrict__ pos,
                              float* __restrict__ H) {
    int bidx = blockIdx.x, tid = threadIdx.x;
    if (bidx == 0) {
        for (int e = tid; e < EMBED; e += 256) H[e] = cls[e] + pos[e];
    } else {
        int row = 2048 + bidx;  // rows 2049..4095
        if (row < S_PAD)
            for (int e = tid; e < EMBED; e += 256) H[(size_t)row * EMBED + e] = 0.f;
    }
}

// ---------------- patch embed: 16 patches per block ----------------
__global__ __launch_bounds__(256) void patch_embed_kernel(
    const float* __restrict__ x, const float* __restrict__ cw, const float* __restrict__ cb,
    const float* __restrict__ pos, float* __restrict__ H) {
    __shared__ __bf16 px[16][1024];
    const int tid = threadIdx.x;
    for (int c = tid; c < 16 * 1024; c += 256) {
        int pl = c >> 10, j = c & 1023;
        int pg = blockIdx.x * 16 + pl;               // patch 0..2047
        int d0 = (pg >> 8) * 4, h0 = ((pg >> 4) & 15) * 16, w0 = (pg & 15) * 16;
        int kd = j >> 8, kh = (j >> 4) & 15, kw = j & 15;
        px[pl][j] = (__bf16)x[(size_t)(d0 + kd) * 65536 + (size_t)(h0 + kh) * 256 + (w0 + kw)];
    }
    __syncthreads();
    float acc[3][16];
#pragma unroll
    for (int oi = 0; oi < 3; ++oi)
#pragma unroll
        for (int pl = 0; pl < 16; ++pl) acc[oi][pl] = 0.f;
    for (int j = 0; j < 1024; ++j) {
        float w0v = cw[(size_t)tid * 1024 + j];
        float w1v = cw[(size_t)(tid + 256) * 1024 + j];
        float w2v = cw[(size_t)(tid + 512) * 1024 + j];
#pragma unroll
        for (int pl = 0; pl < 16; ++pl) {
            float pv = (float)px[pl][j];
            acc[0][pl] += w0v * pv;
            acc[1][pl] += w1v * pv;
            acc[2][pl] += w2v * pv;
        }
    }
#pragma unroll
    for (int oi = 0; oi < 3; ++oi) {
        int o = tid + oi * 256;
#pragma unroll
        for (int pl = 0; pl < 16; ++pl) {
            int row = blockIdx.x * 16 + pl + 1;  // token index (cls at 0)
            H[(size_t)row * EMBED + o] = acc[oi][pl] + cb[o] + pos[(size_t)row * EMBED + o];
        }
    }
}

// ---------------- LayerNorm (rows of h -> bf16) ----------------
__global__ __launch_bounds__(256) void ln_bf16_kernel(
    const float* __restrict__ X, const float* __restrict__ w, const float* __restrict__ b,
    __bf16* __restrict__ out) {
    const int row = blockIdx.x, tid = threadIdx.x;
    const float* xr = X + (size_t)row * EMBED;
    float v0 = xr[tid], v1 = xr[tid + 256], v2 = xr[tid + 512];
    __shared__ float rs[256], rq[256];
    rs[tid] = v0 + v1 + v2;
    rq[tid] = v0 * v0 + v1 * v1 + v2 * v2;
    __syncthreads();
    for (int o = 128; o > 0; o >>= 1) {
        if (tid < o) { rs[tid] += rs[tid + o]; rq[tid] += rq[tid + o]; }
        __syncthreads();
    }
    float mean = rs[0] * (1.f / 768.f);
    float var  = rq[0] * (1.f / 768.f) - mean * mean;
    float inv  = rsqrtf(var + 1e-5f);
    __bf16* o0 = out + (size_t)row * EMBED;
    o0[tid]       = (__bf16)((v0 - mean) * inv * w[tid] + b[tid]);
    o0[tid + 256] = (__bf16)((v1 - mean) * inv * w[tid + 256] + b[tid + 256]);
    o0[tid + 512] = (__bf16)((v2 - mean) * inv * w[tid + 512] + b[tid + 512]);
}

// ---------------- final LayerNorm on row 0 -> fp32 out ----------------
__global__ __launch_bounds__(256) void final_ln_kernel(
    const float* __restrict__ X, const float* __restrict__ w, const float* __restrict__ b,
    float* __restrict__ out) {
    const int tid = threadIdx.x;
    float v0 = X[tid], v1 = X[tid + 256], v2 = X[tid + 512];
    __shared__ float rs[256], rq[256];
    rs[tid] = v0 + v1 + v2;
    rq[tid] = v0 * v0 + v1 * v1 + v2 * v2;
    __syncthreads();
    for (int o = 128; o > 0; o >>= 1) {
        if (tid < o) { rs[tid] += rs[tid + o]; rq[tid] += rq[tid + o]; }
        __syncthreads();
    }
    float mean = rs[0] * (1.f / 768.f);
    float var  = rq[0] * (1.f / 768.f) - mean * mean;
    float inv  = rsqrtf(var + 1e-5f);
    out[tid]       = (v0 - mean) * inv * w[tid] + b[tid];
    out[tid + 256] = (v1 - mean) * inv * w[tid + 256] + b[tid + 256];
    out[tid + 512] = (v2 - mean) * inv * w[tid + 512] + b[tid + 512];
}

// ---------------- WMMA GEMM: C[M,N] = A[M,K](bf16) x B[K,N](bf16) + bias ----------------
// Block tile 128x128, 8 waves (each: 16 rows x 128 cols). K-step 32.
// Double-buffered LDS: A tile via TDM async load (padded rows 32->40 halves in HW),
// B tile via register-staged pipelined loads (transposed into LDS).
// EPI 0: store bf16; EPI 1: store bf16(gelu); EPI 2: fp32 residual accumulate.
template <int EPI>
__global__ __launch_bounds__(256) void gemm_bf16_kernel(
    const __bf16* __restrict__ A, const __bf16* __restrict__ Bw,
    const float* __restrict__ bias, __bf16* __restrict__ Cbf,
    float* __restrict__ Cres, int N, int K) {
    __shared__ __bf16 As[2][128][40];
    __shared__ __bf16 BT[2][128][40];
    const int tid = threadIdx.x, lane = tid & 31, wv = tid >> 5;
    const int n0 = blockIdx.x * 128, m0 = blockIdx.y * 128;
    v8f acc[8];
#pragma unroll
    for (int t = 0; t < 8; ++t) acc[t] = v8f_zero();

    const int KT = K >> 5;

    // B tile: 32(K) x 128(N) halves; 16 halves / thread as 2 x uint4
    auto loadB = [&](int kt, uint4 (&breg)[2]) {
        int k0 = kt << 5;
#pragma unroll
        for (int i = 0; i < 2; ++i) {
            int c = tid * 2 + i;
            int kk = c >> 4, nn = (c & 15) << 3;
            breg[i] = *reinterpret_cast<const uint4*>(Bw + (size_t)(k0 + kk) * N + n0 + nn);
        }
    };
    auto storeB = [&](int buf, uint4 (&breg)[2]) {
#pragma unroll
        for (int i = 0; i < 2; ++i) {
            int c = tid * 2 + i;
            int kk = c >> 4, nn = (c & 15) << 3;
            const __bf16* cp = reinterpret_cast<const __bf16*>(&breg[i]);
#pragma unroll
            for (int j = 0; j < 8; ++j) BT[buf][nn + j][kk] = cp[j];
        }
    };
#ifndef USE_TDM
    auto loadA = [&](int kt, uint4 (&areg)[2]) {
        int k0 = kt << 5;
#pragma unroll
        for (int i = 0; i < 2; ++i) {
            int c = tid * 2 + i;
            int row = c >> 2, koff = (c & 3) << 3;
            areg[i] = *reinterpret_cast<const uint4*>(A + (size_t)(m0 + row) * K + k0 + koff);
        }
    };
    auto storeA = [&](int buf, uint4 (&areg)[2]) {
#pragma unroll
        for (int i = 0; i < 2; ++i) {
            int c = tid * 2 + i;
            int row = c >> 2, koff = (c & 3) << 3;
            *reinterpret_cast<uint4*>(&As[buf][row][koff]) = areg[i];
        }
    };
    uint4 areg[2];
#endif
    uint4 breg[2];

    // prologue: stage tile 0
#ifdef USE_TDM
    if (wv == 0)
        tdm_load_2d(A + (size_t)m0 * K, &As[0][0][0], 32, 128, (unsigned long long)K, 3, 3);
#else
    loadA(0, areg);
    storeA(0, areg);
#endif
    loadB(0, breg);
    storeB(0, breg);
#ifdef USE_TDM
    if (wv == 0) tdm_wait();
#endif
    __syncthreads();

    int cur = 0;
    for (int kt = 0; kt < KT; ++kt) {
        int nxt = cur ^ 1;
        bool more = (kt + 1 < KT);
        if (more) {
#ifdef USE_TDM
            if (wv == 0)
                tdm_load_2d(A + (size_t)m0 * K + ((size_t)(kt + 1) << 5), &As[nxt][0][0],
                            32, 128, (unsigned long long)K, 3, 3);
#else
            loadA(kt + 1, areg);
#endif
            loadB(kt + 1, breg);  // global loads in flight during WMMA below
        }
        // compute on current buffer
        v16bf a = load_a_frag(&As[cur][0][0], 40, wv * 16, 0, lane);
#pragma unroll
        for (int t = 0; t < 8; ++t) {
            v16bf bb = load_b_frag(&BT[cur][0][0], 40, t * 16, 0, lane);
            acc[t] = wmma_bf16(a, bb, acc[t]);
        }
        if (more) {
#ifndef USE_TDM
            storeA(nxt, areg);
#endif
            storeB(nxt, breg);
#ifdef USE_TDM
            if (wv == 0) tdm_wait();
#endif
        }
        __syncthreads();
        cur = nxt;
    }

    // epilogue; C layout: lane<16 -> row g, lane>=16 -> row g+8; col = lane&15
#pragma unroll
    for (int t = 0; t < 8; ++t) {
        int col = n0 + t * 16 + (lane & 15);
        float bv = bias[col];
#pragma unroll
        for (int g = 0; g < 8; ++g) {
            int row = m0 + wv * 16 + ((lane & 16) ? g + 8 : g);
            float v = acc[t][g] + bv;
            if (EPI == 0) {
                Cbf[(size_t)row * N + col] = (__bf16)v;
            } else if (EPI == 1) {
                Cbf[(size_t)row * N + col] = (__bf16)(0.5f * v * (1.f + erff(v * 0.70710678f)));
            } else {
                Cres[(size_t)row * N + col] += v;
            }
        }
    }
}

// ---------------- dilated attention: one block per (segment, head) ----------------
struct AttnShared {
    union {
        struct { __bf16 q[128][72]; __bf16 k[128][72]; } qk;  // 36864 B
        __bf16 P[128][136];                                   // 34816 B (overlays q/k)
    } u;
    __bf16 vT[64][136];                                       // 17408 B
};

__global__ __launch_bounds__(256) void attn_kernel(
    const __bf16* __restrict__ Q, const __bf16* __restrict__ K, const __bf16* __restrict__ V,
    float* __restrict__ Oc, float* __restrict__ Lc) {
    __shared__ AttnShared sm;
    const int tid = threadIdx.x, lane = tid & 31, wv = tid >> 5;
    const int segidx = blockIdx.x / HEADS, h = blockIdx.x % HEADS;
    int b = 0;
    while (b < 4 && segidx >= d_cum[b + 1]) ++b;
    const int w = d_segw[b], r = d_ratio[b];
    const int n = segidx - d_cum[b];
    const int hr = h % r;
    const int base_pos = n * w + hr;
    const size_t gbase = (size_t)base_pos * EMBED + h * DH;

    // gather dilated rows: q/k via TDM strided-2D descriptors (row stride r*768 halves,
    // 64-half rows padded to 72 in LDS); v transposed manually.
#ifdef USE_TDM
    if (wv == 0) {
        tdm_load_2d(Q + gbase, &sm.u.qk.q[0][0], 64, 128, (unsigned long long)r * EMBED, 4, 3);
        tdm_load_2d(K + gbase, &sm.u.qk.k[0][0], 64, 128, (unsigned long long)r * EMBED, 4, 3);
    }
    for (int c = tid; c < 1024; c += 256) {
        int j = c >> 3, d0 = (c & 7) << 3;
        uint4 vv = *reinterpret_cast<const uint4*>(V + gbase + (size_t)(r * j) * EMBED + d0);
        const __bf16* vp = reinterpret_cast<const __bf16*>(&vv);
#pragma unroll
        for (int i = 0; i < 8; ++i) sm.vT[d0 + i][j] = vp[i];
    }
    if (wv == 0) tdm_wait();
#else
    for (int c = tid; c < 1024; c += 256) {
        int j = c >> 3, d0 = (c & 7) << 3;
        size_t goff = gbase + (size_t)(r * j) * EMBED + d0;
        *reinterpret_cast<uint4*>(&sm.u.qk.q[j][d0]) = *reinterpret_cast<const uint4*>(Q + goff);
        *reinterpret_cast<uint4*>(&sm.u.qk.k[j][d0]) = *reinterpret_cast<const uint4*>(K + goff);
        uint4 vv = *reinterpret_cast<const uint4*>(V + goff);
        const __bf16* vp = reinterpret_cast<const __bf16*>(&vv);
#pragma unroll
        for (int i = 0; i < 8; ++i) sm.vT[d0 + i][j] = vp[i];
    }
#endif
    __syncthreads();

    // S = q k^T : each wave owns 16 rows x 128 cols (8 tiles), K=64
    v8f acc[8];
#pragma unroll
    for (int t = 0; t < 8; ++t) acc[t] = v8f_zero();
#pragma unroll
    for (int ko = 0; ko < 64; ko += 32) {
        v16bf a = load_a_frag(&sm.u.qk.q[0][0], 72, wv * 16, ko, lane);
#pragma unroll
        for (int t = 0; t < 8; ++t) {
            v16bf bb = load_b_frag(&sm.u.qk.k[0][0], 72, t * 16, ko, lane);
            acc[t] = wmma_bf16(a, bb, acc[t]);
        }
    }

    // scale + key mask + row softmax (16-lane groups own distinct row sets)
    float mval[8];
#pragma unroll
    for (int t = 0; t < 8; ++t) {
        int col = t * 16 + (lane & 15);
        mval[t] = (base_pos + r * col < N_REAL) ? 0.f : -1e9f;
    }
    float rmax[8], rsum[8];
#pragma unroll
    for (int g = 0; g < 8; ++g) {
        float m = -3.4e38f;
#pragma unroll
        for (int t = 0; t < 8; ++t) {
            float s = acc[t][g] * 0.125f + mval[t];
            acc[t][g] = s;
            m = fmaxf(m, s);
        }
#pragma unroll
        for (int o = 1; o < 16; o <<= 1) m = fmaxf(m, __shfl_xor(m, o, 32));
        float ssum = 0.f;
#pragma unroll
        for (int t = 0; t < 8; ++t) {
            float p = expf(acc[t][g] - m);
            acc[t][g] = p;
            ssum += p;
        }
#pragma unroll
        for (int o = 1; o < 16; o <<= 1) ssum += __shfl_xor(ssum, o, 32);
        rmax[g] = m;
        rsum[g] = ssum;
    }

    __syncthreads();  // all waves done reading q/k before P overlays them

    // write unnormalized probs P (bf16) and lse
#pragma unroll
    for (int g = 0; g < 8; ++g) {
        int row = wv * 16 + ((lane & 16) ? g + 8 : g);
#pragma unroll
        for (int t = 0; t < 8; ++t)
            sm.u.P[row][t * 16 + (lane & 15)] = (__bf16)acc[t][g];
    }
    if ((lane & 15) == 0) {
        size_t lbase = ((size_t)segidx * HEADS + h) * 128;
#pragma unroll
        for (int g = 0; g < 8; ++g) {
            int row = wv * 16 + ((lane & 16) ? g + 8 : g);
            Lc[lbase + row] = rmax[g] + logf(rsum[g]);
        }
    }

    // O = P v : wave's 16 rows x 64 cols, K=128 (wave reads only rows it wrote)
    v8f acc2[4];
#pragma unroll
    for (int t = 0; t < 4; ++t) acc2[t] = v8f_zero();
#pragma unroll
    for (int ko = 0; ko < 128; ko += 32) {
        v16bf a = load_a_frag(&sm.u.P[0][0], 136, wv * 16, ko, lane);
#pragma unroll
        for (int t = 0; t < 4; ++t) {
            v16bf bb = load_b_frag(&sm.vT[0][0], 136, t * 16, ko, lane);
            acc2[t] = wmma_bf16(a, bb, acc2[t]);
        }
    }
    size_t obase = ((size_t)segidx * HEADS + h) * (128 * DH);
#pragma unroll
    for (int g = 0; g < 8; ++g) {
        int row = wv * 16 + ((lane & 16) ? g + 8 : g);
        float inv = 1.f / rsum[g];
#pragma unroll
        for (int t = 0; t < 4; ++t)
            Oc[obase + (size_t)row * DH + t * 16 + (lane & 15)] = acc2[t][g] * inv;
    }
}

// ---------------- branch combine -> ctx (bf16) ----------------
__global__ __launch_bounds__(256) void combine_kernel(
    const float* __restrict__ Oc, const float* __restrict__ Lc, __bf16* __restrict__ ctx) {
    __shared__ float L5[5][HEADS];
    __shared__ float W5[5][HEADS];
    const int p = blockIdx.x, tid = threadIdx.x;
    if (tid < 5 * HEADS) {
        int b = tid / HEADS, h = tid % HEADS;
        int w = d_segw[b], r = d_ratio[b];
        int nn = p / w, offp = p % w, hr = h % r;
        float lv = -1e9f;
        if (offp % r == hr) {
            int j = (offp - hr) / r;
            lv = Lc[((size_t)(d_cum[b] + nn) * HEADS + h) * 128 + j];
        }
        L5[b][h] = lv;
    }
    __syncthreads();
    if (tid < HEADS) {
        int h = tid;
        float m = L5[0][h];
#pragma unroll
        for (int b = 1; b < 5; ++b) m = fmaxf(m, L5[b][h]);
        float e[5], s = 0.f;
#pragma unroll
        for (int b = 0; b < 5; ++b) { e[b] = expf(L5[b][h] - m); s += e[b]; }
        float inv = 1.f / s;
#pragma unroll
        for (int b = 0; b < 5; ++b) W5[b][h] = e[b] * inv;
    }
    __syncthreads();
    for (int e0 = tid; e0 < EMBED; e0 += 256) {
        int h = e0 >> 6, d = e0 & 63;
        float out = 0.f;
#pragma unroll
        for (int b = 0; b < 5; ++b) {
            int w = d_segw[b], r = d_ratio[b];
            int nn = p / w, offp = p % w, hr = h % r;
            if (offp % r == hr) {
                float wt = W5[b][h];
                if (wt != 0.f) {
                    int j = (offp - hr) / r;
                    out += wt * Oc[(((size_t)(d_cum[b] + nn) * HEADS + h) * 128 + j) * DH + d];
                }
            }
        }
        ctx[(size_t)p * EMBED + e0] = (__bf16)out;
    }
}

// ---------------- launch ----------------
extern "C" void kernel_launch(void* const* d_in, const int* in_sizes, int n_in,
                              void* d_out, int out_size, void* d_ws, size_t ws_size,
                              hipStream_t stream) {
    const float* x      = (const float*)d_in[0];
    const float* conv_w = (const float*)d_in[1];
    const float* conv_b = (const float*)d_in[2];
    const float* cls    = (const float*)d_in[3];
    const float* pos    = (const float*)d_in[4];
    const float* ln1w   = (const float*)d_in[5];
    const float* ln1b   = (const float*)d_in[6];
    const float* wq     = (const float*)d_in[7];
    const float* bq     = (const float*)d_in[8];
    const float* wk     = (const float*)d_in[9];
    const float* bk     = (const float*)d_in[10];
    const float* wvv    = (const float*)d_in[11];
    const float* bv     = (const float*)d_in[12];
    const float* wo     = (const float*)d_in[13];
    const float* bo     = (const float*)d_in[14];
    const float* ln2w   = (const float*)d_in[15];
    const float* ln2b   = (const float*)d_in[16];
    const float* w1     = (const float*)d_in[17];
    const float* b1     = (const float*)d_in[18];
    const float* w2     = (const float*)d_in[19];
    const float* b2     = (const float*)d_in[20];
    const float* lnfw   = (const float*)d_in[21];
    const float* lnfb   = (const float*)d_in[22];

    char* ws = (char*)d_ws;
    size_t off = 0;
    auto alloc = [&](size_t bytes) {
        size_t o = off;
        off += (bytes + 255) & ~(size_t)255;
        return o;
    };
    const size_t WQKVO = (size_t)6 * EMBED * EMBED;     // per-tensor total (6 layers)
    const size_t WMLP  = (size_t)6 * EMBED * MLP_DIM;

    __bf16* wq_bf = (__bf16*)(ws + alloc(WQKVO * 2));
    __bf16* wk_bf = (__bf16*)(ws + alloc(WQKVO * 2));
    __bf16* wv_bf = (__bf16*)(ws + alloc(WQKVO * 2));
    __bf16* wo_bf = (__bf16*)(ws + alloc(WQKVO * 2));
    __bf16* w1_bf = (__bf16*)(ws + alloc(WMLP * 2));
    __bf16* w2_bf = (__bf16*)(ws + alloc(WMLP * 2));
    float*  hbuf  = (float*)(ws + alloc((size_t)S_PAD * EMBED * 4));
    __bf16* xln   = (__bf16*)(ws + alloc((size_t)S_PAD * EMBED * 2));
    __bf16* qb    = (__bf16*)(ws + alloc((size_t)S_PAD * EMBED * 2));
    __bf16* kb    = (__bf16*)(ws + alloc((size_t)S_PAD * EMBED * 2));
    __bf16* vb    = (__bf16*)(ws + alloc((size_t)S_PAD * EMBED * 2));
    __bf16* ctx   = (__bf16*)(ws + alloc((size_t)S_PAD * EMBED * 2));
    __bf16* mid   = (__bf16*)(ws + alloc((size_t)S_PAD * MLP_DIM * 2));
    float*  Oc    = (float*)(ws + alloc((size_t)62 * HEADS * 128 * DH * 4));
    float*  Lc    = (float*)(ws + alloc((size_t)62 * HEADS * 128 * 4));

    // weight conversion fp32 -> bf16
    {
        long n1 = (long)WQKVO, n2 = (long)WMLP;
        int g1 = (int)((n1 + 255) / 256), g2 = (int)((n2 + 255) / 256);
        cvt_bf16_kernel<<<g1, 256, 0, stream>>>(wq, wq_bf, n1);
        cvt_bf16_kernel<<<g1, 256, 0, stream>>>(wk, wk_bf, n1);
        cvt_bf16_kernel<<<g1, 256, 0, stream>>>(wvv, wv_bf, n1);
        cvt_bf16_kernel<<<g1, 256, 0, stream>>>(wo, wo_bf, n1);
        cvt_bf16_kernel<<<g2, 256, 0, stream>>>(w1, w1_bf, n2);
        cvt_bf16_kernel<<<g2, 256, 0, stream>>>(w2, w2_bf, n2);
    }

    // h = [cls+pos0; patches+pos; zeros]
    init_h_kernel<<<2048, 256, 0, stream>>>(cls, pos, hbuf);
    patch_embed_kernel<<<128, 256, 0, stream>>>(x, conv_w, conv_b, pos, hbuf);

    const dim3 gemm_grid_e(EMBED / 128, S_PAD / 128);     // 6 x 32
    const dim3 gemm_grid_m(MLP_DIM / 128, S_PAD / 128);   // 24 x 32

    for (int l = 0; l < 6; ++l) {
        const size_t wqo = (size_t)l * EMBED * EMBED;
        const size_t wmo = (size_t)l * EMBED * MLP_DIM;
        // LN1 -> xln
        ln_bf16_kernel<<<S_PAD, 256, 0, stream>>>(hbuf, ln1w + l * EMBED, ln1b + l * EMBED, xln);
        // q, k, v projections
        gemm_bf16_kernel<0><<<gemm_grid_e, 256, 0, stream>>>(xln, wq_bf + wqo, bq + l * EMBED, qb, nullptr, EMBED, EMBED);
        gemm_bf16_kernel<0><<<gemm_grid_e, 256, 0, stream>>>(xln, wk_bf + wqo, bk + l * EMBED, kb, nullptr, EMBED, EMBED);
        gemm_bf16_kernel<0><<<gemm_grid_e, 256, 0, stream>>>(xln, wv_bf + wqo, bv + l * EMBED, vb, nullptr, EMBED, EMBED);
        // dilated attention (62 segments x 12 heads)
        attn_kernel<<<62 * HEADS, 256, 0, stream>>>(qb, kb, vb, Oc, Lc);
        // branch combine -> ctx
        combine_kernel<<<S_PAD, 256, 0, stream>>>(Oc, Lc, ctx);
        // output projection, residual into h
        gemm_bf16_kernel<2><<<gemm_grid_e, 256, 0, stream>>>(ctx, wo_bf + wqo, bo + l * EMBED, nullptr, hbuf, EMBED, EMBED);
        // LN2 -> xln
        ln_bf16_kernel<<<S_PAD, 256, 0, stream>>>(hbuf, ln2w + l * EMBED, ln2b + l * EMBED, xln);
        // MLP
        gemm_bf16_kernel<1><<<gemm_grid_m, 256, 0, stream>>>(xln, w1_bf + wmo, b1 + l * MLP_DIM, mid, nullptr, MLP_DIM, EMBED);
        gemm_bf16_kernel<2><<<gemm_grid_e, 256, 0, stream>>>(mid, w2_bf + wmo, b2 + l * EMBED, nullptr, hbuf, EMBED, MLP_DIM);
    }

    final_ln_kernel<<<1, 256, 0, stream>>>(hbuf, lnfw, lnfb, (float*)d_out);
}
